// MultiHeadedAttention_79946521247894
// MI455X (gfx1250) — matrix operand order repaired and checked by reading, not compiled, not run on hardware
//
#include <hip/hip_runtime.h>
#include <hip/hip_bf16.h>

#define DI __device__ __forceinline__

typedef __bf16 bf16_t;
typedef __attribute__((ext_vector_type(16))) __bf16 v16bf;
typedef __attribute__((ext_vector_type(8)))  __bf16 v8bf;
typedef __attribute__((ext_vector_type(8)))  float  v8f;

namespace cfg {
constexpr int B = 4, S = 2048, E = 1024, H = 16, D = 64;
constexpr int N = B * S;  // 8192 token rows
}

DI bf16_t f2bf(float x) {
  union { float f; unsigned u; } a; a.f = x;
  unsigned r = (a.u + 0x7FFFu + ((a.u >> 16) & 1u)) >> 16;  // RNE
  union { unsigned short s; bf16_t b; } y; y.s = (unsigned short)r;
  return y.b;
}

// A-matrix 16x32 bf16 fragment (ISA 7.12.2): lanes 0-15 row M=lane hold
// K={0..7,16..23}; lanes 16-31 hold K={8..15,24..31}. Two 16B loads per lane.
DI v16bf frag_a(const bf16_t* __restrict__ base, int stride, int lane) {
  const int m  = lane & 15;
  const int kb = (lane >> 4) << 3;  // 0 or 8
  const bf16_t* p = base + (size_t)m * stride + kb;
  union { v16bf v; v8bf h[2]; } u;
  u.h[0] = *(const v8bf*)(p);
  u.h[1] = *(const v8bf*)(p + 16);
  return u.v;
}

// B-matrix 32x16 bf16 fragment: lanes 0-15 col N=lane hold K={0..7,16..23},
// lanes 16-31 hold K={8..15,24..31}. Source contiguous along K, 'stride' between cols.
DI v16bf frag_b(const bf16_t* __restrict__ base, int stride, int lane) {
  const int n  = lane & 15;
  const int kb = (lane >> 4) << 3;  // 0 or 8
  const bf16_t* p = base + (size_t)n * stride + kb;
  union { v16bf v; v8bf h[2]; } u;
  u.h[0] = *(const v8bf*)(p);
  u.h[1] = *(const v8bf*)(p + 16);
  return u.v;
}

DI v8f wmma_bf16(v16bf a, v16bf b, v8f c) {
  return __builtin_amdgcn_wmma_f32_16x16x32_bf16(
      /*neg_a=*/false, a, /*neg_b=*/false, b,
      /*c_mod=*/(short)0, c, /*reuse_a=*/false, /*reuse_b=*/false);
}

// Async global->LDS 16B copy (ASYNCcnt path). Low 32 bits of a generic
// pointer into the LDS aperture are the LDS byte address (ISA 10.2).
DI void async_g2l_b128(unsigned lds_addr, const bf16_t* gaddr) {
  asm volatile("global_load_async_to_lds_b128 %0, %1, off"
               :: "v"(lds_addr), "v"(gaddr) : "memory");
}
DI void wait_async0() {
  asm volatile("s_wait_asynccnt 0x0" ::: "memory");
}

// ---------------------------------------------------------------- convert
__global__ void k_cvt_f32_bf16(const float* __restrict__ s,
                               bf16_t* __restrict__ d, int n) {
  for (int i = blockIdx.x * blockDim.x + threadIdx.x; i < n;
       i += gridDim.x * blockDim.x)
    d[i] = f2bf(s[i]);
}

// -------------------------------------------------------------- QKV GEMM
// Y = X (N x E, bf16) @ W^T. One wave owns a 32(M) x 64(N) tile.
// mode 0: dst[((b*H+h)*S + s)*D + d]   (Q, K)
// mode 1: dst[((b*H+h)*D + d)*S + s]   (V^T)
__global__ __launch_bounds__(256) void k_gemm_qkv(
    const bf16_t* __restrict__ X, const bf16_t* __restrict__ W,
    bf16_t* __restrict__ dst, int mode) {
  using namespace cfg;
  const int lane = threadIdx.x & 31;
  const int wave = threadIdx.x >> 5;
  const int row0 = blockIdx.y * 256 + wave * 32;
  const int col0 = blockIdx.x * 64;

  v8f acc[2][4] = {};
  for (int k0 = 0; k0 < E; k0 += 32) {
    __builtin_prefetch(X + (size_t)row0 * E + k0 + 64, 0, 0);
    v16bf a0 = frag_a(X + (size_t)row0 * E + k0, E, lane);
    v16bf a1 = frag_a(X + (size_t)(row0 + 16) * E + k0, E, lane);
#pragma unroll
    for (int j = 0; j < 4; ++j) {
      v16bf b = frag_b(W + (size_t)(col0 + j * 16) * E + k0, E, lane);
      acc[0][j] = wmma_bf16(a0, b, acc[0][j]);
      acc[1][j] = wmma_bf16(a1, b, acc[1][j]);
    }
  }

  const int hi = lane >> 4, nn = lane & 15;
#pragma unroll
  for (int i = 0; i < 2; ++i)
#pragma unroll
    for (int j = 0; j < 4; ++j)
#pragma unroll
      for (int r = 0; r < 8; ++r) {
        const int M = row0 + i * 16 + r + 8 * hi;   // token row
        const int C = col0 + j * 16 + nn;           // feature col
        const int bb = M >> 11, ss = M & (S - 1);
        const int hh = C >> 6,  dd = C & (D - 1);
        const bf16_t v = f2bf(acc[i][j][r]);
        if (mode == 0)
          dst[((size_t)(bb * H + hh) * S + ss) * D + dd] = v;
        else
          dst[((size_t)(bb * H + hh) * D + dd) * S + ss] = v;
      }
}

// ---------------------------------------------------------- flash attention
// Block = 8 waves, all on the same (b,h) head, 8 consecutive 16-query tiles.
// Per 32-key step the block async-copies the shared K tile (32x64) and V^T
// tile (64x32) into LDS once (ASYNCcnt), then every wave reads its WMMA
// fragments from LDS: 8x global-traffic reuse.
__global__ __launch_bounds__(256) void k_flash_attn(
    const bf16_t* __restrict__ Q, const bf16_t* __restrict__ K,
    const bf16_t* __restrict__ Vt, bf16_t* __restrict__ ctx) {
  using namespace cfg;
  __shared__ __attribute__((aligned(16))) bf16_t Kl[32][64];      // [key][d]
  __shared__ __attribute__((aligned(16))) bf16_t Vl[64][32];      // [d][key]
  __shared__ __attribute__((aligned(16))) bf16_t pbuf[8][16][32]; // per-wave P

  const int tid  = threadIdx.x;
  const int lane = tid & 31;
  const int wave = tid >> 5;
  const int job  = blockIdx.x * 8 + wave;   // B*H*(S/16) jobs
  const int qi   = job & 127;               // 128 query tiles per head
  const int bh   = job >> 7;                // same for all waves in block
  const int s0   = qi * 16;

  const bf16_t* Qh = Q  + (size_t)bh * S * D;
  const bf16_t* Kh = K  + (size_t)bh * S * D;
  const bf16_t* Vh = Vt + (size_t)bh * D * S;

  const v16bf qa0 = frag_a(Qh + (size_t)s0 * D,      D, lane);  // d 0..31
  const v16bf qa1 = frag_a(Qh + (size_t)s0 * D + 32, D, lane);  // d 32..63

  // Per-thread 16B async staging slots (constant across iterations):
  // K tile: thread t -> key row t/8, d col (t%8)*8
  const bf16_t* kg = Kh + (size_t)(tid >> 3) * D + (tid & 7) * 8;
  const unsigned klds = (unsigned)(uintptr_t)&Kl[tid >> 3][(tid & 7) * 8];
  // V tile: thread t -> d row t/4, key col (t%4)*8
  const bf16_t* vg = Vh + (size_t)(tid >> 2) * S + (tid & 3) * 8;
  const unsigned vlds = (unsigned)(uintptr_t)&Vl[tid >> 2][(tid & 3) * 8];

  v8f ot[4] = {};
  float mrow[8], lrow[8];
#pragma unroll
  for (int r = 0; r < 8; ++r) { mrow[r] = -3.0e38f; lrow[r] = 0.0f; }
  const float scale = 0.125f;  // 1/sqrt(64)
  const int hi = lane >> 4, nn = lane & 15;

  for (int t0 = 0; t0 < S; t0 += 32) {
    // cooperative async stage of K/V tiles for keys [t0, t0+32)
    async_g2l_b128(klds, kg + (size_t)t0 * D);
    async_g2l_b128(vlds, vg + t0);
    wait_async0();
    __syncthreads();

    // scores: two 16-key C tiles, 4 wmma, B-frags from LDS
    v8f sc0 = {}, sc1 = {};
    {
      v16bf kb;
      kb = frag_b(&Kl[0][0],       D, lane);  sc0 = wmma_bf16(qa0, kb, sc0);
      kb = frag_b(&Kl[0][0]  + 32, D, lane);  sc0 = wmma_bf16(qa1, kb, sc0);
      kb = frag_b(&Kl[16][0],      D, lane);  sc1 = wmma_bf16(qa0, kb, sc1);
      kb = frag_b(&Kl[16][0] + 32, D, lane);  sc1 = wmma_bf16(qa1, kb, sc1);
    }
    // online softmax per C-row (16-lane groups share a row)
#pragma unroll
    for (int r = 0; r < 8; ++r) {
      float c0 = sc0[r] * scale, c1 = sc1[r] * scale;
      float cm = fmaxf(c0, c1);
      cm = fmaxf(cm, __shfl_xor(cm, 1, 32));
      cm = fmaxf(cm, __shfl_xor(cm, 2, 32));
      cm = fmaxf(cm, __shfl_xor(cm, 4, 32));
      cm = fmaxf(cm, __shfl_xor(cm, 8, 32));
      const float mnew = fmaxf(mrow[r], cm);
      const float corr = __expf(mrow[r] - mnew);
      const float p0 = __expf(c0 - mnew);
      const float p1 = __expf(c1 - mnew);
      float ps = p0 + p1;
      ps += __shfl_xor(ps, 1, 32);
      ps += __shfl_xor(ps, 2, 32);
      ps += __shfl_xor(ps, 4, 32);
      ps += __shfl_xor(ps, 8, 32);
      lrow[r] = lrow[r] * corr + ps;
      mrow[r] = mnew;
#pragma unroll
      for (int t = 0; t < 4; ++t) ot[t][r] *= corr;
      const int M = r + 8 * hi;
      pbuf[wave][M][nn]      = f2bf(p0);
      pbuf[wave][M][nn + 16] = f2bf(p1);
    }
    // P (16x32) @ V (32x64): P re-layout via per-wave LDS, V frags from LDS
    const v16bf pa = frag_a(&pbuf[wave][0][0], 32, lane);
#pragma unroll
    for (int t = 0; t < 4; ++t) {
      v16bf vb = frag_b(&Vl[t * 16][0], 32, lane);
      ot[t] = wmma_bf16(pa, vb, ot[t]);
    }
    __syncthreads();  // protect Kl/Vl before next iteration's async writes
  }

  // normalize and write ctx as bf16 [B,S,E] (merged heads)
  const int b = bh >> 4, h = bh & 15;
#pragma unroll
  for (int r = 0; r < 8; ++r) {
    const float inv = 1.0f / lrow[r];
    const int srow = s0 + r + 8 * hi;
    const size_t rowbase = ((size_t)b * S + srow) * E + (size_t)h * D;
#pragma unroll
    for (int t = 0; t < 4; ++t)
      ctx[rowbase + t * 16 + nn] = f2bf(ot[t][r] * inv);
  }
}

// ----------------------------------------------------- output projection
__global__ __launch_bounds__(256) void k_gemm_out(
    const bf16_t* __restrict__ Xc, const bf16_t* __restrict__ W,
    const float* __restrict__ bias, float* __restrict__ out) {
  using namespace cfg;
  const int lane = threadIdx.x & 31;
  const int wave = threadIdx.x >> 5;
  const int row0 = blockIdx.y * 256 + wave * 32;
  const int col0 = blockIdx.x * 64;

  v8f acc[2][4] = {};
  for (int k0 = 0; k0 < E; k0 += 32) {
    __builtin_prefetch(Xc + (size_t)row0 * E + k0 + 64, 0, 0);
    v16bf a0 = frag_a(Xc + (size_t)row0 * E + k0, E, lane);
    v16bf a1 = frag_a(Xc + (size_t)(row0 + 16) * E + k0, E, lane);
#pragma unroll
    for (int j = 0; j < 4; ++j) {
      v16bf b = frag_b(W + (size_t)(col0 + j * 16) * E + k0, E, lane);
      acc[0][j] = wmma_bf16(a0, b, acc[0][j]);
      acc[1][j] = wmma_bf16(a1, b, acc[1][j]);
    }
  }

  const int hi = lane >> 4, nn = lane & 15;
#pragma unroll
  for (int i = 0; i < 2; ++i)
#pragma unroll
    for (int j = 0; j < 4; ++j)
#pragma unroll
      for (int r = 0; r < 8; ++r) {
        const int M = row0 + i * 16 + r + 8 * hi;
        const int C = col0 + j * 16 + nn;
        out[(size_t)M * E + C] = acc[i][j][r] + bias[C];
      }
}

// ------------------------------------------------------------------ launch
extern "C" void kernel_launch(void* const* d_in, const int* in_sizes, int n_in,
                              void* d_out, int out_size, void* d_ws, size_t ws_size,
                              hipStream_t stream) {
  using namespace cfg;
  const float* X  = (const float*)d_in[0];
  const float* Wq = (const float*)d_in[1];
  const float* Wk = (const float*)d_in[2];
  const float* Wv = (const float*)d_in[3];
  const float* Wo = (const float*)d_in[4];
  const float* bo = (const float*)d_in[5];
  float* out = (float*)d_out;

  char* ws = (char*)d_ws;
  size_t off = 0;
  auto wsalloc = [&](size_t bytes) -> void* {
    void* p = ws + off;
    off += (bytes + 255) & ~(size_t)255;
    return p;
  };
  bf16_t* Xb  = (bf16_t*)wsalloc((size_t)N * E * 2);
  bf16_t* Wqb = (bf16_t*)wsalloc((size_t)E * E * 2);
  bf16_t* Wkb = (bf16_t*)wsalloc((size_t)E * E * 2);
  bf16_t* Wvb = (bf16_t*)wsalloc((size_t)E * E * 2);
  bf16_t* Wob = (bf16_t*)wsalloc((size_t)E * E * 2);
  bf16_t* Qb  = (bf16_t*)wsalloc((size_t)N * E * 2);  // [B,H,S,D]
  bf16_t* Kb  = (bf16_t*)wsalloc((size_t)N * E * 2);  // [B,H,S,D]
  bf16_t* Vtb = (bf16_t*)wsalloc((size_t)N * E * 2);  // [B,H,D,S]
  bf16_t* Cb  = (bf16_t*)wsalloc((size_t)N * E * 2);  // [B,S,E]

  k_cvt_f32_bf16<<<2048, 256, 0, stream>>>(X,  Xb,  N * E);
  k_cvt_f32_bf16<<<1024, 256, 0, stream>>>(Wq, Wqb, E * E);
  k_cvt_f32_bf16<<<1024, 256, 0, stream>>>(Wk, Wkb, E * E);
  k_cvt_f32_bf16<<<1024, 256, 0, stream>>>(Wv, Wvb, E * E);
  k_cvt_f32_bf16<<<1024, 256, 0, stream>>>(Wo, Wob, E * E);

  dim3 gg(E / 64, N / 256);
  k_gemm_qkv<<<gg, 256, 0, stream>>>(Xb, Wqb, Qb, 0);
  k_gemm_qkv<<<gg, 256, 0, stream>>>(Xb, Wkb, Kb, 0);
  k_gemm_qkv<<<gg, 256, 0, stream>>>(Xb, Wvb, Vtb, 1);

  k_flash_attn<<<(B * H * (S / 16)) / 8, 256, 0, stream>>>(Qb, Kb, Vtb, Cb);

  k_gemm_out<<<gg, 256, 0, stream>>>(Cb, Wob, bo, out);
}
